// MambaBlock_73796128080204
// MI455X (gfx1250) — compile-verified, matrix operand order checked by reading, not run on hardware
//
#include <hip/hip_runtime.h>
#include <hip/hip_bf16.h>

#define Bn 2
#define Ln 2048
#define Dm 768
#define DNi 1536
#define NSt 16
#define TN (Bn * Ln)   // 4096 tokens

typedef __attribute__((ext_vector_type(16))) __bf16 v16bf;
typedef __attribute__((ext_vector_type(8)))  __bf16 v8bf;
typedef __attribute__((ext_vector_type(8)))  float  v8f;

__device__ __forceinline__ __bf16 f2bf(float f) {
  unsigned u = __builtin_bit_cast(unsigned, f);
  unsigned r = u + 0x7FFFu + ((u >> 16) & 1u);  // round-to-nearest-even
  return __builtin_bit_cast(__bf16, (unsigned short)(r >> 16));
}
__device__ __forceinline__ float bf2f(__bf16 h) {
  unsigned u = ((unsigned)__builtin_bit_cast(unsigned short, h)) << 16;
  return __builtin_bit_cast(float, u);
}

// ---------------- weight conversion / padding ----------------
__global__ void k_cvt(const float* __restrict__ s, __bf16* __restrict__ d, int n) {
  int i = blockIdx.x * blockDim.x + threadIdx.x;
  if (i < n) d[i] = f2bf(s[i]);
}

__global__ void k_pad_dtw(const float* __restrict__ s, __bf16* __restrict__ d) {
  int i = blockIdx.x * blockDim.x + threadIdx.x;
  if (i >= DNi * 64) return;
  int row = i >> 6, col = i & 63;
  d[i] = (col < 48) ? f2bf(s[row * 48 + col])
                    : __builtin_bit_cast(__bf16, (unsigned short)0);
}

__global__ void k_zero16(unsigned short* __restrict__ p, int n) {
  int i = blockIdx.x * blockDim.x + threadIdx.x;
  if (i < n) p[i] = 0;
}

// ---------------- LayerNorm + FiLM -> bf16 ----------------
__global__ void k_ln_film(const float* __restrict__ x, const float* __restrict__ gamma,
                          const float* __restrict__ beta, const float* __restrict__ ln_g,
                          const float* __restrict__ ln_b, __bf16* __restrict__ xn) {
  int t = blockIdx.x;             // token 0..4095
  int b = t / Ln;
  const float* xr = x + (size_t)t * Dm;
  float v[3], s = 0.f, s2 = 0.f;
#pragma unroll
  for (int i = 0; i < 3; i++) {
    v[i] = xr[threadIdx.x + 256 * i];
    s += v[i]; s2 += v[i] * v[i];
  }
#pragma unroll
  for (int off = 16; off > 0; off >>= 1) {
    s  += __shfl_xor(s,  off, 32);
    s2 += __shfl_xor(s2, off, 32);
  }
  __shared__ float ss[8], ss2[8];
  int wid = threadIdx.x >> 5;
  if ((threadIdx.x & 31) == 0) { ss[wid] = s; ss2[wid] = s2; }
  __syncthreads();
  float S = 0.f, S2 = 0.f;
#pragma unroll
  for (int i = 0; i < 8; i++) { S += ss[i]; S2 += ss2[i]; }
  float mu = S * (1.f / 768.f);
  float var = S2 * (1.f / 768.f) - mu * mu;
  float rs = rsqrtf(var + 1e-5f);
#pragma unroll
  for (int i = 0; i < 3; i++) {
    int d = threadIdx.x + 256 * i;
    float y = (v[i] - mu) * rs * ln_g[d] + ln_b[d];
    y = y * gamma[b * Dm + d] + beta[b * Dm + d];
    xn[(size_t)t * Dm + d] = f2bf(y);
  }
}

// ---------------- WMMA GEMM: C[M,N] = A[M,K](bf16) * W[N,K]^T(bf16) ----------------
__device__ __forceinline__ v16bf ldfrag(const __bf16* p) {
  v8bf lo = *(const v8bf*)p;
  v8bf hi = *(const v8bf*)(p + 16);
  v16bf r;
#pragma unroll
  for (int i = 0; i < 8; i++) { r[i] = lo[i]; r[i + 8] = hi[i]; }
  return r;
}

// Wave tile = (16*MT) rows x (16*NT) cols; B fragments reused across MT row-tiles.
// B fragment loaded immediately before its WMMA pair: this keeps register
// pressure below the spill point and lets the compiler software-pipeline the
// K loop (verified: no scratch traffic, rotated loop with early loads).
// EPI 0: in_proj split (x_in fp32 / z bf16); EPI 1: x_proj split (dt_r bf16 pad64 / B / C);
// EPI 2: softplus(c + bias) -> dt fp32;      EPI 3: c + residual -> out fp32.
template <int MT, int NT, int EPI>
__global__ void k_gemm(const __bf16* __restrict__ A, const __bf16* __restrict__ W,
                       int M, int N, int K,
                       float* __restrict__ out_f, __bf16* __restrict__ out_b,
                       const float* __restrict__ extra,
                       float* __restrict__ aux1, float* __restrict__ aux2) {
  int gtid = blockIdx.x * blockDim.x + threadIdx.x;
  int wid  = gtid >> 5;
  int lane = threadIdx.x & 31;
  int ntn  = (N + 16 * NT - 1) / (16 * NT);
  int total = (M / (16 * MT)) * ntn;
  if (wid >= total) return;                 // wave-uniform: EXEC stays all-ones
  int m0 = (wid / ntn) * (16 * MT);
  int n0 = (wid % ntn) * (16 * NT);

  v8f acc[MT][NT];
#pragma unroll
  for (int r = 0; r < MT; r++)
#pragma unroll
    for (int t = 0; t < NT; t++)
#pragma unroll
      for (int i = 0; i < 8; i++) acc[r][t][i] = 0.f;

  int hl = lane & 15;
  int kb = (lane & 16) ? 8 : 0;             // per-ISA 16-bit fragment K split
  const __bf16* ap = A + (size_t)(m0 + hl) * K + kb;
  const __bf16* wp = W + (size_t)(n0 + hl) * K + kb;

  for (int k = 0; k < K; k += 32) {
    v16bf af[MT];
#pragma unroll
    for (int r = 0; r < MT; r++) af[r] = ldfrag(ap + (size_t)r * 16 * K + k);
    if (k + 32 < K) __builtin_prefetch(ap + k + 32, 0, 1);   // global_prefetch_b8
#pragma unroll
    for (int t = 0; t < NT; t++) {
      v16bf bfr = ldfrag(wp + (size_t)t * 16 * K + k);
#pragma unroll
      for (int r = 0; r < MT; r++)
        acc[r][t] = __builtin_amdgcn_wmma_f32_16x16x32_bf16(
            false, af[r], false, bfr, (short)0, acc[r][t], false, false);
    }
  }

  int ro = (lane & 16) ? 8 : 0;             // C/D layout: row = base + i + ro, col = n0 + hl
#pragma unroll
  for (int r = 0; r < MT; r++) {
#pragma unroll
    for (int t = 0; t < NT; t++) {
      int col = n0 + t * 16 + hl;
      if (col >= N) continue;
#pragma unroll
      for (int i = 0; i < 8; i++) {
        int row = m0 + r * 16 + i + ro;
        float c = acc[r][t][i];
        if constexpr (EPI == 0) {
          if (col < DNi) out_f[(size_t)row * DNi + col] = c;
          else           out_b[(size_t)row * DNi + (col - DNi)] = f2bf(c);
        } else if constexpr (EPI == 1) {
          if (col < 48)      out_b[(size_t)row * 64 + col] = f2bf(c);
          else if (col < 64) aux1[(size_t)row * 16 + (col - 48)] = c;
          else               aux2[(size_t)row * 16 + (col - 64)] = c;
        } else if constexpr (EPI == 2) {
          float xv = c + extra[col];
          out_f[(size_t)row * DNi + col] = (xv > 20.f) ? xv : log1pf(__expf(xv));
        } else {
          out_f[(size_t)row * Dm + col] = c + extra[(size_t)row * Dm + col];
        }
      }
    }
  }
}

// ---------------- causal depthwise conv (W=4) + SiLU ----------------
__global__ void k_conv_silu(const float* __restrict__ xin, const float* __restrict__ cw,
                            const float* __restrict__ cb,
                            float* __restrict__ uf, __bf16* __restrict__ ub) {
  int i = blockIdx.x * blockDim.x + threadIdx.x;
  if (i >= TN * DNi) return;
  int d = i % DNi;
  int t = i / DNi;
  int l = t % Ln;
  float acc = cb[d];
#pragma unroll
  for (int j = 0; j < 4; j++) {
    int ll = l - 3 + j;
    if (ll >= 0) acc += cw[d * 4 + j] * xin[(size_t)(t - 3 + j) * DNi + d];
  }
  float s = acc / (1.f + __expf(-acc));
  uf[i] = s;
  ub[i] = f2bf(s);
}

// ---------------- selective scan: one thread per (b,d), 16 states in VGPRs ----------------
// B/C (shared by all channels of a batch) staged per 128-step chunk in LDS.
#define SCHUNK 128
__global__ void k_scan(const float* __restrict__ dt, const float* __restrict__ u,
                       const float* __restrict__ Bm, const float* __restrict__ Cm,
                       const float* __restrict__ A_log, const float* __restrict__ Dp,
                       const __bf16* __restrict__ z_bf, __bf16* __restrict__ y_bf) {
  int d = blockIdx.x * blockDim.x + threadIdx.x;   // channel 0..1535
  int b = blockIdx.y;
  __shared__ float sB[SCHUNK * NSt], sC[SCHUNK * NSt];
  float Ar[NSt], h[NSt];
#pragma unroll
  for (int n = 0; n < NSt; n++) {
    Ar[n] = -__expf(A_log[d * NSt + n]);
    h[n] = 0.f;
  }
  float Dd = Dp[d];
  for (int c0 = 0; c0 < Ln; c0 += SCHUNK) {
    __syncthreads();
#pragma unroll
    for (int i = 0; i < (SCHUNK * NSt) / 256; i++) {
      int j = threadIdx.x + 256 * i;
      size_t g = ((size_t)(b * Ln + c0)) * NSt + j;
      sB[j] = Bm[g];
      sC[j] = Cm[g];
    }
    __syncthreads();
    for (int lc = 0; lc < SCHUNK; lc++) {
      size_t tt = (size_t)(b * Ln + c0 + lc);
      float dtv = dt[tt * DNi + d];
      float uv  = u[tt * DNi + d];
      float du  = dtv * uv;
      float y   = Dd * uv;
#pragma unroll
      for (int n = 0; n < NSt; n++) {
        float dA = __expf(dtv * Ar[n]);
        h[n] = h[n] * dA + du * sB[lc * NSt + n];
        y += h[n] * sC[lc * NSt + n];
      }
      float zv = bf2f(z_bf[tt * DNi + d]);
      y *= zv / (1.f + __expf(-zv));
      y_bf[tt * DNi + d] = f2bf(y);
    }
  }
}

// ---------------- host orchestration ----------------
extern "C" void kernel_launch(void* const* d_in, const int* in_sizes, int n_in,
                              void* d_out, int out_size, void* d_ws, size_t ws_size,
                              hipStream_t stream) {
  (void)in_sizes; (void)n_in; (void)out_size; (void)ws_size;
  const float* x      = (const float*)d_in[0];
  const float* gamma  = (const float*)d_in[1];
  const float* beta   = (const float*)d_in[2];
  const float* ln_g   = (const float*)d_in[3];
  const float* ln_b   = (const float*)d_in[4];
  const float* in_w   = (const float*)d_in[5];
  const float* conv_w = (const float*)d_in[6];
  const float* conv_b = (const float*)d_in[7];
  const float* xp_w   = (const float*)d_in[8];
  const float* dt_w   = (const float*)d_in[9];
  const float* dt_b   = (const float*)d_in[10];
  const float* A_log  = (const float*)d_in[11];
  const float* Dp     = (const float*)d_in[12];
  const float* out_w  = (const float*)d_in[13];

  char* ws = (char*)d_ws;
  size_t off = 0;
  auto take = [&](size_t bytes) {
    char* p = ws + off;
    off = (off + bytes + 255) & ~(size_t)255;
    return p;
  };

  __bf16* xn_bf  = (__bf16*)take((size_t)TN * Dm * 2);
  __bf16* w_in   = (__bf16*)take((size_t)2 * DNi * Dm * 2);
  __bf16* w_xp   = (__bf16*)take((size_t)80 * DNi * 2);
  __bf16* w_dt   = (__bf16*)take((size_t)DNi * 64 * 2);
  __bf16* w_out  = (__bf16*)take((size_t)Dm * DNi * 2);
  float*  xin_f  = (float*) take((size_t)TN * DNi * 4);
  __bf16* z_bf   = (__bf16*)take((size_t)TN * DNi * 2);
  float*  u_f    = (float*) take((size_t)TN * DNi * 4);
  __bf16* u_bf   = (__bf16*)take((size_t)TN * DNi * 2);
  __bf16* dtr_bf = (__bf16*)take((size_t)TN * 64 * 2);
  float*  Bm_f   = (float*) take((size_t)TN * NSt * 4);
  float*  Cm_f   = (float*) take((size_t)TN * NSt * 4);
  float*  dt_f   = (float*) take((size_t)TN * DNi * 4);
  __bf16* y_bf   = (__bf16*)take((size_t)TN * DNi * 2);

  auto cdiv = [](int a, int b) { return (a + b - 1) / b; };
  auto gemm_blocks = [&](int M, int N, int MT, int NT) {
    int waves = (M / (16 * MT)) * cdiv(N, 16 * NT);
    return cdiv(waves * 32, 256);
  };

  // weight conversion
  k_cvt<<<cdiv(2 * DNi * Dm, 256), 256, 0, stream>>>(in_w,  w_in,  2 * DNi * Dm);
  k_cvt<<<cdiv(80 * DNi,     256), 256, 0, stream>>>(xp_w,  w_xp,  80 * DNi);
  k_cvt<<<cdiv(Dm * DNi,     256), 256, 0, stream>>>(out_w, w_out, Dm * DNi);
  k_pad_dtw<<<cdiv(DNi * 64, 256), 256, 0, stream>>>(dt_w, w_dt);
  k_zero16<<<cdiv(TN * 64, 256), 256, 0, stream>>>((unsigned short*)dtr_bf, TN * 64);

  // LN + FiLM
  k_ln_film<<<TN, 256, 0, stream>>>(x, gamma, beta, ln_g, ln_b, xn_bf);

  // in_proj: [4096,768] x [3072,768]^T -> split x_in (fp32) / z (bf16)
  k_gemm<2, 4, 0><<<gemm_blocks(TN, 2 * DNi, 2, 4), 256, 0, stream>>>(
      xn_bf, w_in, TN, 2 * DNi, Dm, xin_f, z_bf, nullptr, nullptr, nullptr);

  // depthwise conv + SiLU
  k_conv_silu<<<cdiv(TN * DNi, 256), 256, 0, stream>>>(xin_f, conv_w, conv_b, u_f, u_bf);

  // x_proj: [4096,1536] x [80,1536]^T -> dt_r (bf16, pad 64) / B / C (fp32)
  k_gemm<2, 5, 1><<<gemm_blocks(TN, 80, 2, 5), 256, 0, stream>>>(
      u_bf, w_xp, TN, 80, DNi, nullptr, dtr_bf, nullptr, Bm_f, Cm_f);

  // dt_proj + softplus: [4096,64] x [1536,64]^T -> dt fp32
  k_gemm<2, 4, 2><<<gemm_blocks(TN, DNi, 2, 4), 256, 0, stream>>>(
      dtr_bf, w_dt, TN, DNi, 64, dt_f, nullptr, dt_b, nullptr, nullptr);

  // selective scan + gate (B/C staged through LDS per 128-step chunk)
  k_scan<<<dim3(DNi / 256, Bn), 256, 0, stream>>>(
      dt_f, u_f, Bm_f, Cm_f, A_log, Dp, z_bf, y_bf);

  // out_proj + residual: [4096,1536] x [768,1536]^T + x -> out fp32
  k_gemm<2, 4, 3><<<gemm_blocks(TN, Dm, 2, 4), 256, 0, stream>>>(
      y_bf, w_out, TN, Dm, DNi, (float*)d_out, nullptr, x, nullptr, nullptr);
}
